// fePAM_40295383171513
// MI455X (gfx1250) — matrix operand order repaired and checked by reading, not compiled, not run on hardware
//
#include <hip/hip_runtime.h>
#include <math.h>

typedef __attribute__((ext_vector_type(16))) _Float16 v16h;
typedef __attribute__((ext_vector_type(8)))  float    v8f;

#define BB   4
#define CC   64
#define HH   64
#define WW   64
#define KK   81
#define HW   (HH * WW)     // 4096
#define KPAD 96            // 81 candidates padded to 6 tiles of 16

// One wave (32 lanes) per query position. 8 waves / 256-thread block.
__global__ __launch_bounds__(256) void fePAM_attn_wmma(
    const float* __restrict__ Q,
    const float* __restrict__ S,
    const float* __restrict__ R,
    const int*   __restrict__ xxs,
    const int*   __restrict__ yys,
    float* __restrict__ out_buf,   // [B, C, H, W]
    float* __restrict__ out_M)     // [B, HW, K]
{
    __shared__ int   s_lin[8][KPAD];  // gathered linear spatial indices (padded with 0)
    __shared__ float s_q  [8][CC];    // this query's Q vector
    __shared__ float s_sc [8][KPAD];  // scores, then softmax probabilities

    const int lane = threadIdx.x & 31;
    const int wv   = threadIdx.x >> 5;            // wave in block
    const int q    = blockIdx.x * 8 + wv;         // global query id [0, 16384)
    const int b    = q >> 12;                     // / 4096
    const int hw   = q & (HW - 1);

    const float* Sb = S + (size_t)b * CC * HW;
    const float* Rb = R + (size_t)b * CC * HW;
    const float* Qb = Q + (size_t)b * CC * HW;

    // ---- stage indices and Q vector into LDS ----
    {
        const int ib = q * KK;
        const int k2 = lane + 64;
        s_lin[wv][lane]      = xxs[ib + lane]      * WW + yys[ib + lane];
        s_lin[wv][lane + 32] = xxs[ib + lane + 32] * WW + yys[ib + lane + 32];
        s_lin[wv][k2]        = (k2 < KK) ? (xxs[ib + k2] * WW + yys[ib + k2]) : 0;
        s_q[wv][lane]        = Qb[(size_t)lane * HW + hw];
        s_q[wv][lane + 32]   = Qb[(size_t)(lane + 32) * HW + hw];
    }
    __syncthreads();

    const int half = lane >> 4;   // 0: lanes 0-15, 1: lanes 16-31
    const int l15  = lane & 15;

    // ==== GEMM 1: scores[k] = sum_c Key[c,k] * Q[c] via v_wmma_f32_16x16x32_f16 ====
    // B-matrix = Q replicated across the 16 columns.
    // 16-bit B 32x16 layout: lanes 0-15 hold K{0..7,16..23}, lanes 16-31 hold K{8..15,24..31}.
    v16h bq0, bq1;
    #pragma unroll
    for (int j = 0; j < 16; ++j) {
        bq0[j] = (_Float16)s_q[wv][8 * half + j + (j & 8)];        // channels 0..31
        bq1[j] = (_Float16)s_q[wv][32 + 8 * half + j + (j & 8)];   // channels 32..63
    }

    #pragma unroll
    for (int t = 0; t < 6; ++t) {
        const int m   = 16 * t + l15;      // candidate this lane gathers
        const int lin = s_lin[wv][m];
        v8f acc = {};
        // A 16x32 f16 layout: lanes 0-15: halfs = K{0..7,16..23}; lanes 16-31: K{8..15,24..31}
        v16h a;
        #pragma unroll
        for (int j = 0; j < 8; ++j) {
            a[j]     = (_Float16)Sb[(size_t)(8 * half + j)      * HW + lin];
            a[8 + j] = (_Float16)Sb[(size_t)(16 + 8 * half + j) * HW + lin];
        }
        acc = __builtin_amdgcn_wmma_f32_16x16x32_f16(false, a, false, bq0, (short)0, acc, false, false);
        #pragma unroll
        for (int j = 0; j < 8; ++j) {
            a[j]     = (_Float16)Sb[(size_t)(32 + 8 * half + j) * HW + lin];
            a[8 + j] = (_Float16)Sb[(size_t)(48 + 8 * half + j) * HW + lin];
        }
        acc = __builtin_amdgcn_wmma_f32_16x16x32_f16(false, a, false, bq1, (short)0, acc, false, false);
        // C/D layout: lane n<16 holds D[r][n], lane n+16 holds D[8+r][n]; columns identical.
        if (l15 == 0) {
            #pragma unroll
            for (int r = 0; r < 8; ++r)
                s_sc[wv][16 * t + 8 * half + r] = acc[r];
        }
    }
    __syncthreads();

    // ==== softmax over 81 candidates (f32, wave32 shuffle reductions) ====
    const bool  v2  = (lane + 64) < KK;
    const float sc0 = s_sc[wv][lane];
    const float sc1 = s_sc[wv][lane + 32];
    const float sc2 = v2 ? s_sc[wv][lane + 64] : -INFINITY;
    float mx = fmaxf(sc0, fmaxf(sc1, sc2));
    #pragma unroll
    for (int off = 16; off >= 1; off >>= 1)
        mx = fmaxf(mx, __shfl_xor(mx, off, 32));
    float e0 = __expf(sc0 - mx);
    float e1 = __expf(sc1 - mx);
    float e2 = v2 ? __expf(sc2 - mx) : 0.0f;
    float sum = e0 + e1 + e2;
    #pragma unroll
    for (int off = 16; off >= 1; off >>= 1)
        sum += __shfl_xor(sum, off, 32);
    const float inv = 1.0f / sum;
    e0 *= inv; e1 *= inv; e2 *= inv;     // e2 == 0 on padded slots
    s_sc[wv][lane]      = e0;            // same-wave LDS ops are in-order
    s_sc[wv][lane + 32] = e1;
    s_sc[wv][lane + 64] = e2;            // zero-fills padding 81..95
    float* Mq = out_M + (size_t)q * KK;
    Mq[lane]      = e0;
    Mq[lane + 32] = e1;
    if (v2) Mq[lane + 64] = e2;
    __syncthreads();

    // ==== GEMM 2: out[c] = sum_k M[k] * Val[k,c] ====
    // A = M replicated across the 16 rows; 3 K-chunks of 32.
    v16h am[3];
    #pragma unroll
    for (int kc = 0; kc < 3; ++kc) {
        #pragma unroll
        for (int j = 0; j < 8; ++j) {
            am[kc][j]     = (_Float16)s_sc[wv][32 * kc + 8 * half + j];
            am[kc][8 + j] = (_Float16)s_sc[wv][32 * kc + 16 + 8 * half + j];
        }
    }

    #pragma unroll
    for (int c0 = 0; c0 < CC; c0 += 16) {
        const int ch = c0 + l15;                         // this lane's output channel column
        const float* Rch = Rb + (size_t)ch * HW;
        v8f acc = {};
        #pragma unroll
        for (int kc = 0; kc < 3; ++kc) {
            v16h bv;  // B 32x16: lanes 0-15 K{0..7,16..23}, lanes 16-31 K{8..15,24..31} of this chunk
            #pragma unroll
            for (int j = 0; j < 16; ++j) {
                const int k = 32 * kc + 8 * half + j + (j & 8);
                bv[j] = (_Float16)Rch[s_lin[wv][k]];
            }
            acc = __builtin_amdgcn_wmma_f32_16x16x32_f16(false, am[kc], false, bv, (short)0, acc, false, false);
        }
        // D row 0 = output; lane n (n<16) holds channel c0+n in acc[0].
        if (half == 0)
            out_buf[((size_t)b * CC + ch) * HW + hw] = acc[0];
    }
}

extern "C" void kernel_launch(void* const* d_in, const int* in_sizes, int n_in,
                              void* d_out, int out_size, void* d_ws, size_t ws_size,
                              hipStream_t stream) {
    (void)in_sizes; (void)n_in; (void)d_ws; (void)ws_size; (void)out_size;
    const float* Q   = (const float*)d_in[0];
    const float* S   = (const float*)d_in[1];
    const float* R   = (const float*)d_in[2];
    const int*   xxs = (const int*)d_in[3];
    const int*   yys = (const int*)d_in[4];
    float* out_buf = (float*)d_out;                          // [B,C,H,W]
    float* out_M   = out_buf + (size_t)BB * CC * HH * WW;    // [B,HW,K]

    const int n_query = BB * HH * WW;        // 16384
    dim3 grid(n_query / 8), block(256);      // 8 waves (queries) per block
    fePAM_attn_wmma<<<grid, block, 0, stream>>>(Q, S, R, xxs, yys, out_buf, out_M);
}
